// Model_48095043780736
// MI455X (gfx1250) — matrix operand order repaired and checked by reading, not compiled, not run on hardware
//
#include <hip/hip_runtime.h>
#include <hip/hip_bf16.h>
#include <math.h>

// ---------------------------------------------------------------------------
// Problem constants (from reference)
// ---------------------------------------------------------------------------
#define N_SENT 8192
#define SEQ    100
#define BAGS   256
#define C_OUT  230      // cnn_layers
#define K_CLS  53       // num_classes
#define EMB    60       // 50 + 5 + 5
#define T_OUT  98       // SEQ - 2 (valid conv)

// WMMA tiling for the im2col GEMM:  [98 x 180] x [180 x 230]
#define MT 7            // m-tiles of 16 rows  -> 112 (rows >=98 masked)
#define NT 15           // n-tiles of 16 cols  -> 240 (cols >=230 are pad)
#define KT 6            // k-steps of 32       -> 192 (k >=180 is zero pad)
#define KPAD 192
#define LDS_EMB (112 * EMB + KPAD)   // 6912 halves: flat embedding + zero tail

typedef __attribute__((ext_vector_type(16))) _Float16 v16h;
typedef __attribute__((ext_vector_type(8)))  float    v8f;
typedef __attribute__((ext_vector_type(4)))  _Float16 h4;   // 8-byte aligned

// ---------------------------------------------------------------------------
// Workspace layout (bytes)
// ---------------------------------------------------------------------------
#define WS_PACKB   0                                   // 46080 halves = 92160 B
#define WS_SENT    (92160)                             // 8192*230 f32
#define WS_E       (WS_SENT + N_SENT * C_OUT * 4)      // 8192*53 f32
#define WS_ALPHA   (WS_E + N_SENT * K_CLS * 4)         // 8192*53 f32
#define WS_BAG     (WS_ALPHA + N_SENT * K_CLS * 4)     // 256*53*230 f32

// ---------------------------------------------------------------------------
// Kernel 0: pack conv weights into WMMA B-fragment layout (f16).
// B is [K=192 x N=240]; fragment (kt, nt) stored as [32 lanes][16 halves]:
//   lanes 0-15 : element i -> k = kt*32 + i (i<8) or kt*32 + 8 + i (i>=8), n = nt*16 + lane
//   lanes16-31 : element i -> k = kt*32 + 8 + i (i<8) or kt*32 + 16 + i (i>=8), n = nt*16 + lane-16
// ---------------------------------------------------------------------------
__global__ __launch_bounds__(256) void prepack_kernel(
    const float* __restrict__ conv_w, _Float16* __restrict__ packB) {
  int i = blockIdx.x * 256 + threadIdx.x;          // over KT*NT*32*16 = 46080
  if (i >= KT * NT * 32 * 16) return;
  int e    = i & 15;
  int lane = (i >> 4) & 31;
  int frag = i >> 9;
  int nt = frag % NT, kt = frag / NT;
  int k = kt * 32 + ((lane >> 4) ? 8 : 0) + (e & 7) + ((e >> 3) ? 16 : 0);
  int c = nt * 16 + (lane & 15);
  float v = 0.f;
  if (k < 3 * EMB && c < C_OUT) v = conv_w[c * (3 * EMB) + k];  // (C,1,3,60)
  packB[i] = (_Float16)v;
}

// ---------------------------------------------------------------------------
// Kernel 1: fused embedding-gather + conv + max-over-time + tanh.
// One block per sentence; 7 waves, each wave owns one 16-row m-tile.
// ---------------------------------------------------------------------------
__global__ __launch_bounds__(224) void conv_fused_kernel(
    const int* __restrict__ x, const int* __restrict__ ldist,
    const int* __restrict__ rdist,
    const float* __restrict__ Wv, const float* __restrict__ pf1,
    const float* __restrict__ pf2, const float* __restrict__ conv_b,
    const _Float16* __restrict__ packB, float* __restrict__ sent) {
  __shared__ _Float16 embf[LDS_EMB];           // flat [100*60] + zero tail
  __shared__ float colmax[MT][NT * 16];        // per-m-tile column maxima

  const int n   = blockIdx.x;
  const int tid = threadIdx.x;

  // -------- gather embeddings (f32 tables -> f16 LDS, flattened) ----------
  for (int i = tid; i < SEQ * EMB; i += 224) {
    int l = i / EMB, d = i % EMB;
    float v;
    if (d < 50)      v = Wv[(long)x[n * SEQ + l] * 50 + d];
    else if (d < 55) v = pf1[ldist[n * SEQ + l] * 5 + (d - 50)];
    else             v = pf2[rdist[n * SEQ + l] * 5 + (d - 55)];
    embf[i] = (_Float16)v;
  }
  for (int i = SEQ * EMB + tid; i < LDS_EMB; i += 224) embf[i] = (_Float16)0.f;
  __syncthreads();

  const int wave  = tid >> 5;          // m-tile id, 0..6
  const int lane  = tid & 31;
  const int khalf = lane >> 4;         // 0: lanes 0-15, 1: lanes 16-31
  const int mrow  = wave * 16 + (lane & 15);   // im2col row this lane feeds

  // -------- build the 6 A fragments (held in registers) -------------------
  // A[m, k] = embf[m*60 + k]; per-lane halves come in two aligned 8-half runs.
  v16h afrag[KT];
  const h4* lds4 = (const h4*)embf;
  const int hbase = mrow * (EMB / 4);          // m*60 halves = m*15 h4s
#pragma unroll
  for (int kt = 0; kt < KT; ++kt) {
    const int kb   = kt * 32;
    const int off0 = khalf ? 8 : 0;            // halves [off0, off0+8)
    const int off1 = khalf ? 24 : 16;          // halves [off1, off1+8)
    h4 q0 = lds4[hbase + (kb + off0) / 4];
    h4 q1 = lds4[hbase + (kb + off0) / 4 + 1];
    h4 q2 = lds4[hbase + (kb + off1) / 4];
    h4 q3 = lds4[hbase + (kb + off1) / 4 + 1];
    v16h a;
    a[0]=q0[0]; a[1]=q0[1]; a[2]=q0[2]; a[3]=q0[3];
    a[4]=q1[0]; a[5]=q1[1]; a[6]=q1[2]; a[7]=q1[3];
    a[8]=q2[0]; a[9]=q2[1]; a[10]=q2[2]; a[11]=q2[3];
    a[12]=q3[0]; a[13]=q3[1]; a[14]=q3[2]; a[15]=q3[3];
    afrag[kt] = a;
  }

  // -------- GEMM: each wave sweeps all 15 n-tiles --------------------------
  const v16h* pb = (const v16h*)packB;
  const int mbase = wave * 16 + (khalf ? 8 : 0);   // C/D row base for this lane
  for (int nt = 0; nt < NT; ++nt) {
    v8f acc = {};
#pragma unroll
    for (int kt = 0; kt < KT; ++kt) {
      v16h b = pb[(kt * NT + nt) * 32 + lane];
      acc = __builtin_amdgcn_wmma_f32_16x16x32_f16(
          false, afrag[kt], false, b, (short)0, acc, false, false);
    }
    // max over the 8 output rows this lane holds (mask t >= 98)
    float mx = -INFINITY;
#pragma unroll
    for (int r = 0; r < 8; ++r)
      if (mbase + r < T_OUT) mx = fmaxf(mx, acc[r]);
    // combine the two lane-halves (rows M and M+8 live in lane ^ 16)
    float other = __shfl_xor(mx, 16, 32);
    mx = fmaxf(mx, other);
    if (lane < 16) colmax[wave][nt * 16 + lane] = mx;
  }
  __syncthreads();

  // -------- reduce 7 m-tiles, add bias, tanh, store ------------------------
  for (int c = tid; c < C_OUT; c += 224) {
    float m = colmax[0][c];
#pragma unroll
    for (int w = 1; w < MT; ++w) m = fmaxf(m, colmax[w][c]);
    sent[n * C_OUT + c] = tanhf(m + conv_b[c]);
  }
}

// ---------------------------------------------------------------------------
// Kernel 2: per-sentence logits  e[n,k] = sent[n,:] . rcnn_w[k,:] + rcnn_b[k]
// ---------------------------------------------------------------------------
__global__ __launch_bounds__(64) void e_kernel(
    const float* __restrict__ sent, const float* __restrict__ rw,
    const float* __restrict__ rb, float* __restrict__ e) {
  __shared__ float srow[C_OUT];
  int n = blockIdx.x;
  for (int c = threadIdx.x; c < C_OUT; c += 64) srow[c] = sent[n * C_OUT + c];
  __syncthreads();
  int k = threadIdx.x;
  if (k < K_CLS) {
    float s = rb[k];
    const float* wr = rw + k * C_OUT;
    for (int c = 0; c < C_OUT; ++c) s += srow[c] * wr[c];
    e[n * K_CLS + k] = s;
  }
}

// ---------------------------------------------------------------------------
// Kernel 3: per-bag per-class softmax over sentences -> alpha, then
//           bag[b,k,c] = sum_{n in bag} alpha[n,k] * sent[n,c]
// Bags are contiguous index ranges, so one block per bag.
// ---------------------------------------------------------------------------
__global__ __launch_bounds__(256) void bag_kernel(
    const float* __restrict__ e, const float* __restrict__ sent,
    const int* __restrict__ offsets, float* __restrict__ alpha,
    float* __restrict__ bag) {
  __shared__ float mk[K_CLS], dk[K_CLS];
  const int b  = blockIdx.x;
  const int t0 = offsets[b], t1 = offsets[b + 1];
  const int S  = t1 - t0;
  const int tid = threadIdx.x;

  if (tid < K_CLS) {
    float m = -INFINITY;
    for (int n = t0; n < t1; ++n) m = fmaxf(m, e[n * K_CLS + tid]);
    float d = 0.f;
    for (int n = t0; n < t1; ++n) d += __expf(e[n * K_CLS + tid] - m);
    mk[tid] = m;
    dk[tid] = d;
  }
  __syncthreads();

  for (int i = tid; i < S * K_CLS; i += 256) {
    int n = t0 + i / K_CLS, k = i % K_CLS;
    alpha[n * K_CLS + k] = __expf(e[n * K_CLS + k] - mk[k]) / dk[k];
  }
  __threadfence_block();
  __syncthreads();

  for (int i = tid; i < K_CLS * C_OUT; i += 256) {
    int k = i / C_OUT, c = i % C_OUT;
    float s = 0.f;
    for (int n = t0; n < t1; ++n)
      s += alpha[n * K_CLS + k] * sent[n * C_OUT + c];
    bag[((long)b * K_CLS + k) * C_OUT + c] = s;
  }
}

// ---------------------------------------------------------------------------
// Kernel 4: o[b,k,j] = bag[b,k,:] . rcnn_w[j,:] + rb[j];
//           softmax over j; output diagonal element.
// ---------------------------------------------------------------------------
__global__ __launch_bounds__(256) void out_kernel(
    const float* __restrict__ bag, const float* __restrict__ rw,
    const float* __restrict__ rb, float* __restrict__ out) {
  __shared__ float o[K_CLS * K_CLS];
  const int b = blockIdx.x, tid = threadIdx.x;
  for (int i = tid; i < K_CLS * K_CLS; i += 256) {
    int k = i / K_CLS, j = i % K_CLS;
    const float* br = bag + ((long)b * K_CLS + k) * C_OUT;
    const float* wr = rw + j * C_OUT;
    float s = rb[j];
    for (int c = 0; c < C_OUT; ++c) s += br[c] * wr[c];
    o[i] = s;
  }
  __syncthreads();
  if (tid < K_CLS) {
    float m = -INFINITY;
    for (int j = 0; j < K_CLS; ++j) m = fmaxf(m, o[tid * K_CLS + j]);
    float d = 0.f;
    for (int j = 0; j < K_CLS; ++j) d += __expf(o[tid * K_CLS + j] - m);
    out[b * K_CLS + tid] = __expf(o[tid * K_CLS + tid] - m) / d;
  }
}

// ---------------------------------------------------------------------------
// Launcher
// ---------------------------------------------------------------------------
extern "C" void kernel_launch(void* const* d_in, const int* in_sizes, int n_in,
                              void* d_out, int out_size, void* d_ws,
                              size_t ws_size, hipStream_t stream) {
  const int*   x       = (const int*)d_in[0];
  const int*   ldist   = (const int*)d_in[1];
  const int*   rdist   = (const int*)d_in[2];
  const int*   offsets = (const int*)d_in[3];   // total_shape, BAGS+1
  const float* Wv      = (const float*)d_in[4];
  const float* pf1     = (const float*)d_in[5];
  const float* pf2     = (const float*)d_in[6];
  const float* conv_w  = (const float*)d_in[7];
  const float* conv_b  = (const float*)d_in[8];
  const float* rcnn_w  = (const float*)d_in[9];
  const float* rcnn_b  = (const float*)d_in[10];
  float* out = (float*)d_out;

  char* ws = (char*)d_ws;
  _Float16* packB = (_Float16*)(ws + WS_PACKB);
  float*    sent  = (float*)(ws + WS_SENT);
  float*    e     = (float*)(ws + WS_E);
  float*    alpha = (float*)(ws + WS_ALPHA);
  float*    bag   = (float*)(ws + WS_BAG);

  prepack_kernel<<<(KT * NT * 32 * 16 + 255) / 256, 256, 0, stream>>>(conv_w,
                                                                      packB);
  conv_fused_kernel<<<N_SENT, 224, 0, stream>>>(x, ldist, rdist, Wv, pf1, pf2,
                                                conv_b, packB, sent);
  e_kernel<<<N_SENT, 64, 0, stream>>>(sent, rcnn_w, rcnn_b, e);
  bag_kernel<<<BAGS, 256, 0, stream>>>(e, sent, offsets, alpha, bag);
  out_kernel<<<BAGS, 256, 0, stream>>>(bag, rcnn_w, rcnn_b, out);
}
